// GPTLTP_44538810860195
// MI455X (gfx1250) — matrix operand order repaired and checked by reading, not compiled, not run on hardware
//
#include <hip/hip_runtime.h>
#include <hip/hip_bf16.h>
#include <math.h>

// Model constants (match reference)
#define L_   12
#define H_   12
#define C_   768
#define HS_  64
#define V_   50304
#define B_   2
#define T_   1024
#define C3_  (3 * C_)
#define CF_  (4 * C_)
#define MR_  (B_ * T_)      // 2048 rows of activations
#define MIN_KEEP_ 64

typedef __bf16 bf16x16 __attribute__((ext_vector_type(16)));
typedef __bf16 bf16x8  __attribute__((ext_vector_type(8)));
typedef __bf16 bf16x4  __attribute__((ext_vector_type(4)));
typedef float  f32x8   __attribute__((ext_vector_type(8)));

static __device__ __forceinline__ f32x8 wmma_bf16(bf16x16 a, bf16x16 b, f32x8 c) {
    return __builtin_amdgcn_wmma_f32_16x16x32_bf16(
        false, a, false, b, (short)0, c, false, false);
}

static __device__ __forceinline__ bf16x16 cat8(bf16x8 lo, bf16x8 hi) {
    return __builtin_shufflevector(lo, hi, 0, 1, 2, 3, 4, 5, 6, 7,
                                   8, 9, 10, 11, 12, 13, 14, 15);
}

static __device__ __forceinline__ bf16x8 cat4(bf16x4 lo, bf16x4 hi) {
    return __builtin_shufflevector(lo, hi, 0, 1, 2, 3, 4, 5, 6, 7);
}

static __device__ __forceinline__ bf16x4 cvt4(float4 v) {
    bf16x4 r;
    r[0] = (__bf16)v.x; r[1] = (__bf16)v.y; r[2] = (__bf16)v.z; r[3] = (__bf16)v.w;
    return r;
}

// ---------------------------------------------------------------------------
// bf16-WMMA GEMM: C[M,N] = act(A[M,K] @ B + bias) + resid
// bTransposed==0: B row-major [K,N]; ==1: B row-major [N,K] (used for wte^T).
// Block tile 64x128, 8 waves (wm 0..3, wn 0..1); per wave: 1 A-frag x 4 B-frags
// per K-step of 32. LDS B tile stored transposed so fragments are contiguous.
// Requires M%64==0, N%128==0, K%32==0 (true for all uses here).
// ---------------------------------------------------------------------------
#define KP_ 40   // padded K stride (bf16 elems): 80B, 16B-aligned rows
__global__ __launch_bounds__(256)
void gemm_bf16_kernel(const float* __restrict__ A, const float* __restrict__ Bm,
                      const float* __restrict__ bias, const float* __restrict__ resid,
                      float* __restrict__ Co, int M, int N, int K,
                      int bTransposed, int gelu) {
    __shared__ __bf16 As[64][KP_];     // [m][k]
    __shared__ __bf16 Bst[128][KP_];   // [n][k] (transposed tile)

    const int tid  = threadIdx.x;
    const int lane = tid & 31;
    const int w    = tid >> 5;
    const int wm   = w >> 1;      // 0..3
    const int wn   = w & 1;       // 0..1
    const int row0 = blockIdx.y * 64;
    const int col0 = blockIdx.x * 128;

    f32x8 acc[4] = {};

    const int mloc  = 16 * wm + (lane & 15);
    const int koffA = (lane & 16) ? 8 : 0;   // A-frag K offset (ISA layout)
    const int kbB   = (lane & 16) ? 16 : 0;  // unused; B-frags use koffA (same k order as A)
    const int nbase = 64 * wn + (lane & 15);
    const int rbase = (lane & 16) ? 8 : 0;   // D-frag row base (ISA layout)

    for (int k0 = 0; k0 < K; k0 += 32) {
        __syncthreads();
        // Stage A: 64x32 floats = 512 float4, 2 per thread
#pragma unroll
        for (int j = 0; j < 2; ++j) {
            int i  = tid + 256 * j;
            int r  = i >> 3;
            int cq = (i & 7) * 4;
            float4 v = *(const float4*)&A[(size_t)(row0 + r) * K + (k0 + cq)];
            *(bf16x4*)&As[r][cq] = cvt4(v);
        }
        // Stage B: 32x128 floats = 1024 float4, 4 per thread, into Bst[n][k]
        if (bTransposed) {
#pragma unroll
            for (int j = 0; j < 4; ++j) {
                int i  = tid + 256 * j;
                int n  = i >> 3;
                int kq = (i & 7) * 4;
                float4 v = *(const float4*)&Bm[(size_t)(col0 + n) * K + (k0 + kq)];
                *(bf16x4*)&Bst[n][kq] = cvt4(v);
            }
        } else {
#pragma unroll
            for (int j = 0; j < 4; ++j) {
                int i  = tid + 256 * j;
                int r  = i >> 5;            // k within step
                int cq = (i & 31) * 4;      // n
                float4 v = *(const float4*)&Bm[(size_t)(k0 + r) * N + (col0 + cq)];
                bf16x4 t = cvt4(v);
                Bst[cq + 0][r] = t[0];
                Bst[cq + 1][r] = t[1];
                Bst[cq + 2][r] = t[2];
                Bst[cq + 3][r] = t[3];
            }
        }
        if (k0 + 32 < K && tid < 64) {
            __builtin_prefetch(&A[(size_t)(row0 + tid) * K + (k0 + 32)], 0, 1);
        }
        __syncthreads();

        bf16x8 alo = *(const bf16x8*)&As[mloc][koffA];
        bf16x8 ahi = *(const bf16x8*)&As[mloc][16 + koffA];
        bf16x16 a = cat8(alo, ahi);
#pragma unroll
        for (int t = 0; t < 4; ++t) {
            const __bf16* bp = &Bst[nbase + 16 * t][koffA];
            bf16x16 b = cat8(*(const bf16x8*)bp, *(const bf16x8*)(bp + 16));
            acc[t] = wmma_bf16(a, b, acc[t]);
        }
    }

#pragma unroll
    for (int t = 0; t < 4; ++t) {
#pragma unroll
        for (int r = 0; r < 8; ++r) {
            int m = row0 + 16 * wm + rbase + r;
            int c = col0 + 64 * wn + 16 * t + (lane & 15);
            float v = acc[t][r];
            if (bias) v += bias[c];
            if (gelu) v = 0.5f * v * (1.0f + erff(v * 0.70710678118654752f));
            if (resid) v += resid[(size_t)m * N + c];
            Co[(size_t)m * N + c] = v;
        }
    }
}

// ---------------------------------------------------------------------------
// Fused attention for one (b, h, 16-query tile): S = QK^T/8 (WMMA, bf16),
// causal + key mask, full-row softmax in LDS, importance column sums
// (atomics into imp[b,t]), Y = P@V (WMMA, V staged transposed in LDS).
// ---------------------------------------------------------------------------
__global__ __launch_bounds__(128)
void attn_kernel(const float* __restrict__ qkv, const float* __restrict__ am,
                 float* __restrict__ y, float* __restrict__ imp) {
    __shared__ float  S[16][T_];        // 64 KB score / prob tile
    __shared__ __bf16 Vst[HS_][KP_];    // V tile transposed: [hd][key-in-block]
    __shared__ float  red[16][8];
    __shared__ float  rowM[16], rowL[16];

    const int qb = blockIdx.x & 63;
    const int h  = (blockIdx.x >> 6) % H_;
    const int b  = blockIdx.x / (64 * H_);
    const int tid  = threadIdx.x;
    const int lane = tid & 31;
    const int w    = tid >> 5;

    const int m     = lane & 15;
    const int koff  = (lane & 16) ? 8 : 0;
    const int keyl  = lane & 15;
    const int kdb   = (lane & 16) ? 16 : 0;
    const int rbase = (lane & 16) ? 8 : 0;

    // Q fragments (head-dim 0..31 and 32..63) via vectorized global loads
    bf16x16 a0, a1;
    {
        const float* qp = &qkv[((size_t)(b * T_ + qb * 16 + m)) * C3_ + h * HS_];
        float4 p0 = *(const float4*)(qp + koff);
        float4 p1 = *(const float4*)(qp + koff + 4);
        float4 p2 = *(const float4*)(qp + 16 + koff);
        float4 p3 = *(const float4*)(qp + 16 + koff + 4);
        a0 = cat8(cat4(cvt4(p0), cvt4(p1)), cat4(cvt4(p2), cvt4(p3)));
        float4 q0 = *(const float4*)(qp + 32 + koff);
        float4 q1 = *(const float4*)(qp + 32 + koff + 4);
        float4 q2 = *(const float4*)(qp + 48 + koff);
        float4 q3 = *(const float4*)(qp + 48 + koff + 4);
        a1 = cat8(cat4(cvt4(q0), cvt4(q1)), cat4(cvt4(q2), cvt4(q3)));
    }

    // Scores: each wave handles key-blocks w, w+4, ..., 60
    for (int kb = w; kb < 64; kb += 4) {
        int key = kb * 16 + keyl;
        const float* kp = &qkv[((size_t)(b * T_ + key)) * C3_ + C_ + h * HS_];
        float4 p0 = *(const float4*)(kp + koff);
        float4 p1 = *(const float4*)(kp + koff + 4);
        float4 p2 = *(const float4*)(kp + 16 + koff);
        float4 p3 = *(const float4*)(kp + 16 + koff + 4);
        bf16x16 b0 = cat8(cat4(cvt4(p0), cvt4(p1)), cat4(cvt4(p2), cvt4(p3)));
        float4 q0 = *(const float4*)(kp + 32 + koff);
        float4 q1 = *(const float4*)(kp + 32 + koff + 4);
        float4 q2 = *(const float4*)(kp + 48 + koff);
        float4 q3 = *(const float4*)(kp + 48 + koff + 4);
        bf16x16 b1 = cat8(cat4(cvt4(q0), cvt4(q1)), cat4(cvt4(q2), cvt4(q3)));

        f32x8 z = {};
        f32x8 s = wmma_bf16(a1, b1, wmma_bf16(a0, b0, z));
#pragma unroll
        for (int r = 0; r < 8; ++r) {
            int mr   = rbase + r;
            int qidx = qb * 16 + mr;
            int kidx = kb * 16 + keyl;
            float v = s[r] * 0.125f;  // 1/sqrt(64)
            if (kidx > qidx || am[b * T_ + kidx] <= 0.0f) v = -INFINITY;
            S[mr][kidx] = v;
        }
    }
    __syncthreads();

    // Softmax: 8 threads per row, 128 contiguous cols each (float4)
    {
        int row = tid >> 3, sub = tid & 7;
        float* sr = &S[row][sub * 128];
        float pm = -INFINITY;
        for (int c = 0; c < 128; c += 4) {
            float4 v = *(const float4*)(sr + c);
            pm = fmaxf(pm, fmaxf(fmaxf(v.x, v.y), fmaxf(v.z, v.w)));
        }
        red[row][sub] = pm;
        __syncthreads();
        if (sub == 0) {
            float mx = red[row][0];
            for (int i = 1; i < 8; ++i) mx = fmaxf(mx, red[row][i]);
            rowM[row] = mx;
        }
        __syncthreads();
        float mx = rowM[row];
        int dead = !(mx > -INFINITY);
        float sum = 0.0f;
        for (int c = 0; c < 128; c += 4) {
            float4 v = *(const float4*)(sr + c);
            v.x = dead ? 0.0f : __expf(v.x - mx);
            v.y = dead ? 0.0f : __expf(v.y - mx);
            v.z = dead ? 0.0f : __expf(v.z - mx);
            v.w = dead ? 0.0f : __expf(v.w - mx);
            *(float4*)(sr + c) = v;
            sum += v.x + v.y + v.z + v.w;
        }
        red[row][sub] = sum;
        __syncthreads();
        if (sub == 0) {
            float t = 0.0f;
            for (int i = 0; i < 8; ++i) t += red[row][i];
            rowL[row] = (t > 0.0f) ? (1.0f / t) : 0.0f;
        }
        __syncthreads();
        float inv = rowL[row];
        for (int c = 0; c < 128; c += 4) {
            float4 v = *(const float4*)(sr + c);
            v.x *= inv; v.y *= inv; v.z *= inv; v.w *= inv;
            *(float4*)(sr + c) = v;
        }
    }
    __syncthreads();

    // Importance: column sums of P, accumulated across (h, q) by atomics
    for (int c = tid; c < T_; c += 128) {
        float sum = 0.0f;
#pragma unroll
        for (int r = 0; r < 16; ++r) sum += S[r][c];
        atomicAdd(&imp[b * T_ + c], sum);
    }

    // Y = P @ V ; wave w owns head-dim cols [16w, 16w+16). V staged per 32-key
    // block, transposed in LDS so B-fragments are contiguous ds_load_b128.
    f32x8 acc = {};
    const int hd0 = 16 * w;
    for (int kb = 0; kb < 32; ++kb) {
        __syncthreads();
        // Stage V block: 32 keys x 64 hd = 512 float4, 4 per thread
#pragma unroll
        for (int j = 0; j < 4; ++j) {
            int i   = tid + 128 * j;
            int key = i >> 4;
            int hq  = (i & 15) * 4;
            float4 v = *(const float4*)&qkv[((size_t)(b * T_ + kb * 32 + key)) * C3_
                                            + 2 * C_ + h * HS_ + hq];
            bf16x4 t = cvt4(v);
            Vst[hq + 0][key] = t[0];
            Vst[hq + 1][key] = t[1];
            Vst[hq + 2][key] = t[2];
            Vst[hq + 3][key] = t[3];
        }
        __syncthreads();

        // A fragment from probabilities in S (two contiguous 8-float runs)
        bf16x16 a;
        {
            const float* sp = &S[m][kb * 32];
            float4 p0 = *(const float4*)(sp + koff);
            float4 p1 = *(const float4*)(sp + koff + 4);
            float4 p2 = *(const float4*)(sp + 16 + koff);
            float4 p3 = *(const float4*)(sp + 16 + koff + 4);
            a = cat8(cat4(cvt4(p0), cvt4(p1)), cat4(cvt4(p2), cvt4(p3)));
        }
        const __bf16* vp = &Vst[hd0 + keyl][koff];
        bf16x16 bv = cat8(*(const bf16x8*)vp, *(const bf16x8*)(vp + 16));
        acc = wmma_bf16(a, bv, acc);
    }
#pragma unroll
    for (int r = 0; r < 8; ++r) {
        int q = qb * 16 + rbase + r;
        y[((size_t)(b * T_ + q)) * C_ + h * HS_ + hd0 + keyl] = acc[r];
    }
}

// ---------------------------------------------------------------------------
// LayerNorm: one block per row (C=768)
// ---------------------------------------------------------------------------
__global__ __launch_bounds__(256)
void ln_kernel(const float* __restrict__ x, const float* __restrict__ w,
               const float* __restrict__ bb, float* __restrict__ out) {
    __shared__ float s1[256], s2[256];
    int row = blockIdx.x, tid = threadIdx.x;
    const float* xr = x + (size_t)row * C_;
    float sum = 0.0f, sq = 0.0f;
    for (int c = tid; c < C_; c += 256) { float v = xr[c]; sum += v; sq += v * v; }
    s1[tid] = sum; s2[tid] = sq;
    __syncthreads();
    for (int off = 128; off > 0; off >>= 1) {
        if (tid < off) { s1[tid] += s1[tid + off]; s2[tid] += s2[tid + off]; }
        __syncthreads();
    }
    float mean = s1[0] * (1.0f / C_);
    float var  = s2[0] * (1.0f / C_) - mean * mean;
    float inv  = rsqrtf(var + 1e-5f);
    for (int c = tid; c < C_; c += 256)
        out[(size_t)row * C_ + c] = (xr[c] - mean) * inv * w[c] + bb[c];
}

__global__ void embed_kernel(const int* __restrict__ idx, const float* __restrict__ wte,
                             const float* __restrict__ wpe, float* __restrict__ x) {
    int i = blockIdx.x * blockDim.x + threadIdx.x;   // float4 index
    if (i >= MR_ * C_ / 4) return;
    int c4 = i % (C_ / 4);
    int bt = i / (C_ / 4);
    int t  = bt % T_;
    float4 a = *(const float4*)&wte[(size_t)idx[bt] * C_ + c4 * 4];
    float4 p = *(const float4*)&wpe[(size_t)t * C_ + c4 * 4];
    float4 r; r.x = a.x + p.x; r.y = a.y + p.y; r.z = a.z + p.z; r.w = a.w + p.w;
    *(float4*)&x[(size_t)i * 4] = r;
}

__global__ void fill_kernel(float* p, float v, int n) {
    int i = blockIdx.x * blockDim.x + threadIdx.x;
    if (i < n) p[i] = v;
}

__global__ void prune_kernel(const float* __restrict__ imp, const float* __restrict__ th,
                             float* __restrict__ am, float* __restrict__ cm) {
    int i = blockIdx.x * blockDim.x + threadIdx.x;
    if (i >= MR_) return;
    int t = i % T_;
    float score = imp[i] * (1.0f / (H_ * T_));   // mean over heads and queries
    float pm = (score >= *th) ? 1.0f : 0.0f;
    if (t >= T_ - MIN_KEEP_) pm = 1.0f;          // protected tail
    float c = am[i] * pm;
    am[i] = c;
    cm[i] = c;
}

__global__ void scale_kernel(float* __restrict__ x, const float* __restrict__ cm) {
    int i = blockIdx.x * blockDim.x + threadIdx.x;   // float4 index
    if (i >= MR_ * C_ / 4) return;
    float s = cm[i / (C_ / 4)];
    float4 v = *(const float4*)&x[(size_t)i * 4];
    v.x *= s; v.y *= s; v.z *= s; v.w *= s;
    *(float4*)&x[(size_t)i * 4] = v;
}

// ---------------------------------------------------------------------------
// Per-row NLL over V=50304 (max, logsumexp, gather target)
// ---------------------------------------------------------------------------
__global__ __launch_bounds__(256)
void loss_kernel(const float* __restrict__ logits, const int* __restrict__ targets,
                 float* __restrict__ accum) {
    __shared__ float s1[256];
    int row = blockIdx.x, tid = threadIdx.x;
    const float* lr = logits + (size_t)row * V_;
    float mx = -INFINITY;
    for (int c = tid; c < V_; c += 256) mx = fmaxf(mx, lr[c]);
    s1[tid] = mx;
    __syncthreads();
    for (int off = 128; off > 0; off >>= 1) {
        if (tid < off) s1[tid] = fmaxf(s1[tid], s1[tid + off]);
        __syncthreads();
    }
    float m = s1[0];
    __syncthreads();
    float sum = 0.0f;
    for (int c = tid; c < V_; c += 256) sum += __expf(lr[c] - m);
    s1[tid] = sum;
    __syncthreads();
    for (int off = 128; off > 0; off >>= 1) {
        if (tid < off) s1[tid] += s1[tid + off];
        __syncthreads();
    }
    if (tid == 0) {
        int tgt = targets[row];
        if (tgt != -1) {
            float nll = -(lr[tgt] - m - __logf(s1[0]));
            atomicAdd(&accum[0], nll);
            atomicAdd(&accum[1], 1.0f);
        }
    }
}

__global__ void finalize_kernel(const float* __restrict__ accum, float* __restrict__ out) {
    out[0] = (accum[1] > 0.0f) ? (accum[0] / accum[1]) : 0.0f;
}

// ---------------------------------------------------------------------------
extern "C" void kernel_launch(void* const* d_in, const int* in_sizes, int n_in,
                              void* d_out, int out_size, void* d_ws, size_t ws_size,
                              hipStream_t stream) {
    (void)in_sizes; (void)n_in; (void)out_size; (void)ws_size;
    const int*   idx        = (const int*)d_in[0];
    const int*   targets    = (const int*)d_in[1];
    const float* wte        = (const float*)d_in[2];
    const float* wpe        = (const float*)d_in[3];
    const float* ln1_w      = (const float*)d_in[4];
    const float* ln1_b      = (const float*)d_in[5];
    const float* attn_w     = (const float*)d_in[6];
    const float* attn_b     = (const float*)d_in[7];
    const float* proj_w     = (const float*)d_in[8];
    const float* proj_b     = (const float*)d_in[9];
    const float* ln2_w      = (const float*)d_in[10];
    const float* ln2_b      = (const float*)d_in[11];
    const float* fc_w       = (const float*)d_in[12];
    const float* fc_b       = (const float*)d_in[13];
    const float* mlp_w      = (const float*)d_in[14];
    const float* mlp_b      = (const float*)d_in[15];
    const float* thresholds = (const float*)d_in[16];
    const float* lnf_w      = (const float*)d_in[17];
    const float* lnf_b      = (const float*)d_in[18];

    float* out = (float*)d_out;          // [B*T*V] logits, then [1] loss
    float* ws  = (float*)d_ws;
    size_t off = 0;
    float* x    = ws + off; off += (size_t)MR_ * C_;
    float* hbuf = ws + off; off += (size_t)MR_ * C_;
    float* qkv  = ws + off; off += (size_t)MR_ * C3_;
    float* ybuf = ws + off; off += (size_t)MR_ * C_;
    float* ff   = ws + off; off += (size_t)MR_ * CF_;
    float* imp  = ws + off; off += MR_;
    float* am   = ws + off; off += MR_;
    float* cm   = ws + off; off += MR_;
    float* acc  = ws + off; off += 2;

    embed_kernel<<<(MR_ * C_ / 4 + 255) / 256, 256, 0, stream>>>(idx, wte, wpe, x);
    fill_kernel<<<(MR_ + 255) / 256, 256, 0, stream>>>(am, 1.0f, MR_);
    fill_kernel<<<1, 256, 0, stream>>>(acc, 0.0f, 2);

    for (int l = 0; l < L_; ++l) {
        ln_kernel<<<MR_, 256, 0, stream>>>(x, ln1_w + l * C_, ln1_b + l * C_, hbuf);
        gemm_bf16_kernel<<<dim3(C3_ / 128, MR_ / 64), 256, 0, stream>>>(
            hbuf, attn_w + (size_t)l * C_ * C3_, attn_b + (size_t)l * C3_,
            nullptr, qkv, MR_, C3_, C_, 0, 0);
        fill_kernel<<<(MR_ + 255) / 256, 256, 0, stream>>>(imp, 0.0f, MR_);
        attn_kernel<<<B_ * H_ * 64, 128, 0, stream>>>(qkv, am, ybuf, imp);
        gemm_bf16_kernel<<<dim3(C_ / 128, MR_ / 64), 256, 0, stream>>>(
            ybuf, proj_w + (size_t)l * C_ * C_, proj_b + (size_t)l * C_,
            x, x, MR_, C_, C_, 0, 0);
        prune_kernel<<<(MR_ + 255) / 256, 256, 0, stream>>>(imp, thresholds + l, am, cm);
        scale_kernel<<<(MR_ * C_ / 4 + 255) / 256, 256, 0, stream>>>(x, cm);
        ln_kernel<<<MR_, 256, 0, stream>>>(x, ln2_w + l * C_, ln2_b + l * C_, hbuf);
        gemm_bf16_kernel<<<dim3(CF_ / 128, MR_ / 64), 256, 0, stream>>>(
            hbuf, fc_w + (size_t)l * C_ * CF_, fc_b + (size_t)l * CF_,
            nullptr, ff, MR_, CF_, C_, 0, 1);
        gemm_bf16_kernel<<<dim3(C_ / 128, MR_ / 64), 256, 0, stream>>>(
            ff, mlp_w + (size_t)l * CF_ * C_, mlp_b + (size_t)l * C_,
            x, x, MR_, C_, CF_, 0, 0);
    }

    ln_kernel<<<MR_, 256, 0, stream>>>(x, lnf_w, lnf_b, hbuf);
    // logits = x @ wte^T : wte is [V, C] row-major -> transposed-B mode
    gemm_bf16_kernel<<<dim3(V_ / 128, MR_ / 64), 256, 0, stream>>>(
        hbuf, wte, nullptr, nullptr, out, MR_, V_, C_, 1, 0);
    loss_kernel<<<MR_, 256, 0, stream>>>(out, targets, acc);
    finalize_kernel<<<1, 1, 0, stream>>>(acc, out + (size_t)MR_ * V_);
}
